// Transformer_smallCTMModule_AS_57320633533092
// MI455X (gfx1250) — matrix operand order repaired and checked by reading, not compiled, not run on hardware
//
#include <hip/hip_runtime.h>

// ---------------------------------------------------------------------------
// Types for CDNA5 WMMA (wave32)
// ---------------------------------------------------------------------------
typedef __attribute__((ext_vector_type(16))) __bf16        v16bf;
typedef __attribute__((ext_vector_type(8)))  float         v8f;
typedef __attribute__((ext_vector_type(8)))  unsigned int  v8u;

// ---------------------------------------------------------------------------
// Model constants (from the reference)
// ---------------------------------------------------------------------------
static constexpr int  BATCH = 64;
static constexpr int  NTOK  = 256;    // GRID*GRID
static constexpr int  CD0   = 384;
static constexpr int  CD1   = 768;
static constexpr int  CHID  = 768;
static constexpr int  H0_   = 4;
static constexpr int  H1_   = 8;
static constexpr int  CLN   = 32;     // cluster_num = 256*0.125
static constexpr int  KNN   = 5;
static constexpr int  SR0_  = 8;
static constexpr int  NITER = 2;      // (12-2)//4

// ---------------------------------------------------------------------------
// Device helpers
// ---------------------------------------------------------------------------
__device__ __forceinline__ unsigned short f2bf(float f) {
  unsigned u = __builtin_bit_cast(unsigned, f);
  u += 0x7fffu + ((u >> 16) & 1u);          // round-to-nearest-even
  return (unsigned short)(u >> 16);
}

__device__ __forceinline__ float gelu_tanh(float x) {
  float x3 = x * x * x;
  return 0.5f * x * (1.0f + tanhf(0.7978845608f * (x + 0.044715f * x3)));
}

// ---------------------------------------------------------------------------
// WMMA GEMM:  C[M,N] = act( A[M,K] * B(K,N) + bias )
//  - f32 in HBM, converted to bf16 while staging into LDS; f32 accumulate
//  - block = 256 threads = 8 waves; block tile 128x32; wave tile 16x32
//  - double-buffered LDS: stage tile k+1 before computing tile k (1 barrier/iter)
//  - uniform fast-path staging when the tile is fully in-bounds
//  - rowGather (A row remap) hoisted out of the K loop
// ---------------------------------------------------------------------------
__global__ __launch_bounds__(256)
void wmma_gemm_kernel(const float* __restrict__ A, const float* __restrict__ Bm,
                      float* __restrict__ C, const float* __restrict__ bias,
                      int M, int N, int K,
                      long lda, long ldb, long ldc,
                      int transB, int act, int innerB,
                      long sAo, long sAi, long sBo, long sBi, long sCo, long sCi,
                      const int* __restrict__ rowGather)
{
  __shared__ unsigned short As[2][128][34];   // +2 pad: conflict-free dword reads
  __shared__ unsigned short Bs[2][32][34];

  int z  = blockIdx.z;
  int bo = z / innerB, bi = z - bo * innerB;
  A  += (long)bo * sAo + (long)bi * sAi;
  Bm += (long)bo * sBo + (long)bi * sBi;
  C  += (long)bo * sCo + (long)bi * sCi;

  const int n0  = blockIdx.x * 32;
  const int m0  = blockIdx.y * 128;
  const int tid = threadIdx.x;
  const int lane = tid & 31, wid = tid >> 5;
  const int lm = lane & 15, lh = lane >> 4;

  const int aK    = tid & 31;   // k index this thread stages for A
  const int aRow0 = tid >> 5;   // base row (step 8)

  // ---- hoist A row indices (and rowGather) out of the K loop ----
  int aRowIdx[16];
  unsigned amask = 0;
#pragma unroll
  for (int e = 0; e < 16; e++) {
    int gm = m0 + aRow0 + e * 8;
    int src = 0;
    if (gm < M) { amask |= 1u << e; src = rowGather ? rowGather[gm] : gm; }
    aRowIdx[e] = src;
  }
  const bool mFull = (m0 + 128 <= M);
  const bool nFull = (n0 + 32 <= N);

  v8f acc0 = {0.f, 0.f, 0.f, 0.f, 0.f, 0.f, 0.f, 0.f};
  v8f acc1 = {0.f, 0.f, 0.f, 0.f, 0.f, 0.f, 0.f, 0.f};

  auto stage = [&](int p, int k0) {
    const bool kFull = (k0 + 32 <= K);
    const int  gk    = k0 + aK;
    // ----- A tile 128x32 -----
    if (mFull && kFull) {
#pragma unroll
      for (int e = 0; e < 16; e++)
        As[p][aRow0 + e * 8][aK] = f2bf(A[(long)aRowIdx[e] * lda + gk]);
      if (!rowGather && (k0 + 64 <= K))
        __builtin_prefetch(&A[(long)aRowIdx[0] * lda + gk + 32], 0, 1);
    } else {
#pragma unroll
      for (int e = 0; e < 16; e++) {
        float v = 0.f;
        if (((amask >> e) & 1u) && gk < K) v = A[(long)aRowIdx[e] * lda + gk];
        As[p][aRow0 + e * 8][aK] = f2bf(v);
      }
    }
    // ----- B tile 32x32 into Bs[n][k] -----
    if (nFull && kFull) {
      if (transB) {
#pragma unroll
        for (int e = 0; e < 4; e++) {
          int idx = tid + e * 256;
          int bk = idx & 31, bn = idx >> 5;
          Bs[p][bn][bk] = f2bf(Bm[(long)(n0 + bn) * ldb + (k0 + bk)]);
        }
      } else {
#pragma unroll
        for (int e = 0; e < 4; e++) {
          int idx = tid + e * 256;
          int bk = idx & 31, bn = idx >> 5;
          Bs[p][bn][bk] = f2bf(Bm[(long)(k0 + bk) * ldb + (n0 + bn)]);
        }
      }
    } else {
#pragma unroll
      for (int e = 0; e < 4; e++) {
        int idx = tid + e * 256;
        int bk = idx & 31, bn = idx >> 5;
        int gkk = k0 + bk, gnn = n0 + bn;
        float v = 0.f;
        if (gkk < K && gnn < N)
          v = transB ? Bm[(long)gnn * ldb + gkk] : Bm[(long)gkk * ldb + gnn];
        Bs[p][bn][bk] = f2bf(v);
      }
    }
  };

  auto compute = [&](int p) {
    v8u au, bu0, bu1;
    const unsigned short* arow = &As[p][wid * 16 + lm][0];
#pragma unroll
    for (int j = 0; j < 4; j++) {
      au[j]     = *(const unsigned int*)(arow + (lh * 8 + 2 * j));        // K 0..7 / 8..15
      au[j + 4] = *(const unsigned int*)(arow + (16 + lh * 8 + 2 * j));   // K 16..23 / 24..31
    }
    const unsigned short* br0 = &Bs[p][lm][0];
    const unsigned short* br1 = &Bs[p][16 + lm][0];
#pragma unroll
    for (int j = 0; j < 8; j++) {
      bu0[j] = *(const unsigned int*)(br0 + (lh * 8 + 2 * j + (j >= 4 ? 8 : 0)));   // K 0..7,16..23 / 8..15,24..31
      bu1[j] = *(const unsigned int*)(br1 + (lh * 8 + 2 * j + (j >= 4 ? 8 : 0)));
    }
    acc0 = __builtin_amdgcn_wmma_f32_16x16x32_bf16(
        false, __builtin_bit_cast(v16bf, au),
        false, __builtin_bit_cast(v16bf, bu0),
        (short)0, acc0, false, false);
    acc1 = __builtin_amdgcn_wmma_f32_16x16x32_bf16(
        false, __builtin_bit_cast(v16bf, au),
        false, __builtin_bit_cast(v16bf, bu1),
        (short)0, acc1, false, false);
  };

  stage(0, 0);
  int p = 0;
  for (int k0 = 0; k0 < K; k0 += 32) {
    __syncthreads();
    if (k0 + 32 < K) stage(p ^ 1, k0 + 32);   // overlap next-tile loads with WMMA
    compute(p);
    p ^= 1;
  }

  // ---- epilogue: bias + activation, f32 store (two 16-col halves) ----
#pragma unroll
  for (int r = 0; r < 8; r++) {
    int gm  = m0 + wid * 16 + r + lh * 8;
    int gn0 = n0 + lm;
    int gn1 = n0 + 16 + lm;
    if (gm < M) {
      if (gn0 < N) {
        float v = acc0[r];
        if (bias) v += bias[gn0];
        if (act == 1) v = gelu_tanh(v);
        C[(long)gm * ldc + gn0] = v;
      }
      if (gn1 < N) {
        float v = acc1[r];
        if (bias) v += bias[gn1];
        if (act == 1) v = gelu_tanh(v);
        C[(long)gm * ldc + gn1] = v;
      }
    }
  }
}

// ---------------------------------------------------------------------------
// LayerNorm: one wave per row (wave32 shuffle reductions)
// ---------------------------------------------------------------------------
__global__ __launch_bounds__(256)
void ln_kernel(const float* __restrict__ x, float* __restrict__ y,
               const float* __restrict__ g, const float* __restrict__ b,
               long rows, int C)
{
  int wid = threadIdx.x >> 5, lane = threadIdx.x & 31;
  long row = (long)blockIdx.x * 8 + wid;
  if (row >= rows) return;
  const float* p = x + row * C;
  float s = 0.f, ss = 0.f;
  for (int c = lane; c < C; c += 32) { float v = p[c]; s += v; ss += v * v; }
  for (int o = 16; o; o >>= 1) { s += __shfl_xor(s, o); ss += __shfl_xor(ss, o); }
  float mean = s / C;
  float var  = ss / C - mean * mean;
  float rstd = rsqrtf(var + 1e-5f);
  float* q = y + row * C;
  for (int c = lane; c < C; c += 32)
    q[c] = (p[c] - mean) * rstd * g[c] + b[c];
}

// ---------------------------------------------------------------------------
// Softmax over rows (in-place), scale applied before exp
// ---------------------------------------------------------------------------
__global__ __launch_bounds__(256)
void softmax_kernel(float* __restrict__ a, long rows, int cols, float scale)
{
  int wid = threadIdx.x >> 5, lane = threadIdx.x & 31;
  long row = (long)blockIdx.x * 8 + wid;
  if (row >= rows) return;
  float* p = a + row * (long)cols;
  float mx = -3.4e38f;
  for (int c = lane; c < cols; c += 32) mx = fmaxf(mx, p[c] * scale);
  for (int o = 16; o; o >>= 1) mx = fmaxf(mx, __shfl_xor(mx, o));
  float s = 0.f;
  for (int c = lane; c < cols; c += 32) { float e = __expf(p[c] * scale - mx); p[c] = e; s += e; }
  for (int o = 16; o; o >>= 1) s += __shfl_xor(s, o);
  float inv = 1.f / s;
  for (int c = lane; c < cols; c += 32) p[c] *= inv;
}

// ---------------------------------------------------------------------------
// Elementwise utilities
// ---------------------------------------------------------------------------
__global__ void vadd_kernel(float* __restrict__ x, const float* __restrict__ a, long n) {
  long t = (long)blockIdx.x * blockDim.x + threadIdx.x;
  long stride = (long)gridDim.x * blockDim.x;
  for (; t < n; t += stride) x[t] += a[t];
}
__global__ void addpos_kernel(float* __restrict__ x, const float* __restrict__ pos,
                              long n, long period) {
  long t = (long)blockIdx.x * blockDim.x + threadIdx.x;
  long stride = (long)gridDim.x * blockDim.x;
  for (; t < n; t += stride) x[t] += pos[t % period];
}
__global__ void fz_kernel(float* __restrict__ x, long n) {
  long t = (long)blockIdx.x * blockDim.x + threadIdx.x;
  long stride = (long)gridDim.x * blockDim.x;
  for (; t < n; t += stride) x[t] = 0.f;
}

// ---------------------------------------------------------------------------
// im2col for the 16x16/stride16 patch conv: A[b*256+n, c*256+py*16+px]
// ---------------------------------------------------------------------------
__global__ void im2col_patch_kernel(const float* __restrict__ img, float* __restrict__ out) {
  long total = (long)BATCH * NTOK * 768;
  long t = (long)blockIdx.x * blockDim.x + threadIdx.x;
  long stride = (long)gridDim.x * blockDim.x;
  for (; t < total; t += stride) {
    long r = t / 768;  int k = (int)(t - r * 768);
    int b = (int)(r >> 8), n = (int)(r & 255);
    int gy = n >> 4, gx = n & 15;
    int c = k >> 8, rem = k & 255;
    int py = rem >> 4, px = rem & 15;
    out[t] = img[(((long)(b * 3 + c) * 256 + gy * 16 + py) * 256) + gx * 16 + px];
  }
}

// ---------------------------------------------------------------------------
// im2col for the SR 8x8/stride8 conv on token-major h[B,256,384]:
//   A[b*4 + ty*2+tx, i*64 + ky*8 + kx]
// ---------------------------------------------------------------------------
__global__ void im2col_sr_kernel(const float* __restrict__ h, float* __restrict__ out) {
  long total = (long)BATCH * 4 * (CD0 * 64);
  long t = (long)blockIdx.x * blockDim.x + threadIdx.x;
  long stride = (long)gridDim.x * blockDim.x;
  for (; t < total; t += stride) {
    long r = t / (CD0 * 64); int k = (int)(t - r * (CD0 * 64));
    int b = (int)(r >> 2), tok = (int)(r & 3);
    int ty = tok >> 1, tx = tok & 1;
    int i = k >> 6, rem = k & 63;
    int ky = rem >> 3, kx = rem & 7;
    int n = (ty * 8 + ky) * 16 + (tx * 8 + kx);
    out[t] = h[((long)b * NTOK + n) * CD0 + i];
  }
}

// ---------------------------------------------------------------------------
// CTM kernels (DPC-kNN density clustering + weighted merge)
// ---------------------------------------------------------------------------
__global__ __launch_bounds__(256)
void ctm_prep_kernel(const float* __restrict__ x, const float* __restrict__ sw, const float* __restrict__ sb,
                     const float* __restrict__ sal, float* __restrict__ weight, float* __restrict__ x2,
                     long rows, int C)
{
  int wid = threadIdx.x >> 5, lane = threadIdx.x & 31;
  long row = (long)blockIdx.x * 8 + wid;
  if (row >= rows) return;
  const float* p = x + row * C;
  float d = 0.f, q = 0.f;
  for (int c = lane; c < C; c += 32) { float v = p[c]; d += v * sw[c]; q += v * v; }
  for (int o = 16; o; o >>= 1) { d += __shfl_xor(d, o); q += __shfl_xor(q, o); }
  if (lane == 0) {
    weight[row] = __expf(d + sb[0]) * (sal[row] + 1e-6f);
    x2[row] = q;
  }
}

// gram -> dist (in place): dist = sqrt(max(x2n + x2m - 2g, 0)) / sqrt(C)
__global__ void dist_kernel(float* __restrict__ gram, const float* __restrict__ x2) {
  long total = (long)BATCH * NTOK * NTOK;
  long t = (long)blockIdx.x * blockDim.x + threadIdx.x;
  long stride = (long)gridDim.x * blockDim.x;
  float inv = rsqrtf((float)CD0);
  for (; t < total; t += stride) {
    int b = (int)(t >> 16);
    int n = (int)((t >> 8) & 255);
    int m = (int)(t & 255);
    float sq = x2[b * NTOK + n] + x2[b * NTOK + m] - 2.f * gram[t];
    gram[t] = sqrtf(fmaxf(sq, 0.f)) * inv;
  }
}

__global__ void density_kernel(const float* __restrict__ dist, float* __restrict__ dens) {
  long t = (long)blockIdx.x * blockDim.x + threadIdx.x;
  if (t >= (long)BATCH * NTOK) return;
  const float* row = dist + t * NTOK;
  float best[KNN];
#pragma unroll
  for (int i = 0; i < KNN; i++) best[i] = 3.4e38f;
  for (int m = 0; m < NTOK; m++) {
    float d = row[m];
    int mi = 0; float mv = best[0];
#pragma unroll
    for (int i = 1; i < KNN; i++) if (best[i] > mv) { mv = best[i]; mi = i; }
    if (d < mv) best[mi] = d;
  }
  float s = 0.f;
#pragma unroll
  for (int i = 0; i < KNN; i++) s += best[i] * best[i];
  dens[t] = __expf(-s * (1.0f / KNN));
}

__global__ __launch_bounds__(256)
void dmax_kernel(const float* __restrict__ dist, float* __restrict__ dmax) {
  __shared__ float sm[256];
  int b = blockIdx.x, t = threadIdx.x;
  const float* p = dist + (long)b * NTOK * NTOK;
  float mx = 0.f;
  for (int i = t; i < NTOK * NTOK; i += 256) mx = fmaxf(mx, p[i]);
  sm[t] = mx; __syncthreads();
  for (int s = 128; s; s >>= 1) { if (t < s) sm[t] = fmaxf(sm[t], sm[t + s]); __syncthreads(); }
  if (t == 0) dmax[b] = sm[0];
}

__global__ void dp_kernel(const float* __restrict__ dist, const float* __restrict__ dens,
                          const float* __restrict__ dmax, float* __restrict__ val) {
  long t = (long)blockIdx.x * blockDim.x + threadIdx.x;
  if (t >= (long)BATCH * NTOK) return;
  int b = (int)(t >> 8);
  const float* row = dist + t * NTOK;
  const float* dn  = dens + (long)b * NTOK;
  float my = dens[t];
  float dmin = 3.4e38f;
  for (int m = 0; m < NTOK; m++)
    if (dn[m] > my) dmin = fminf(dmin, row[m]);
  val[t] = fminf(dmin, dmax[b]) * my;
}

// iterative top-32 per batch, one wave per batch
__global__ __launch_bounds__(32)
void top32_kernel(const float* __restrict__ val, int* __restrict__ index_down) {
  int b = blockIdx.x, lane = threadIdx.x;
  float v[8];
#pragma unroll
  for (int j = 0; j < 8; j++) v[j] = val[b * NTOK + lane + 32 * j];
  for (int i = 0; i < CLN; i++) {
    float bv = -3.4e38f; int bi = 0;
#pragma unroll
    for (int j = 0; j < 8; j++) {
      float c = v[j];
      if (c > bv) { bv = c; bi = lane + 32 * j; }
    }
    for (int o = 16; o; o >>= 1) {
      float ov = __shfl_xor(bv, o);
      int   oi = __shfl_xor(bi, o);
      if (ov > bv || (ov == bv && oi < bi)) { bv = ov; bi = oi; }
    }
    if (lane == (bi & 31)) v[bi >> 5] = -3.4e38f;
    if (lane == 0) index_down[b * CLN + i] = bi;
  }
}

__global__ void assign_kernel(const float* __restrict__ dist, const int* __restrict__ index_down,
                              int* __restrict__ idxc) {
  long t = (long)blockIdx.x * blockDim.x + threadIdx.x;
  if (t >= (long)BATCH * NTOK) return;
  int b = (int)(t >> 8), n = (int)(t & 255);
  const int* id = index_down + b * CLN;
  float bd = 3.4e38f; int bj = 0;
  for (int j = 0; j < CLN; j++) {
    float d = dist[((long)b * NTOK + id[j]) * NTOK + n];
    if (d < bd) { bd = d; bj = j; }
  }
  idxc[t] = bj;
}
__global__ void override_kernel(const int* __restrict__ index_down, int* __restrict__ idxc) {
  int t = blockIdx.x * blockDim.x + threadIdx.x;
  if (t >= BATCH * CLN) return;
  int b = t >> 5, j = t & 31;
  idxc[b * NTOK + index_down[t]] = j;
}
__global__ void rowmap_kernel(const int* __restrict__ idxc, int* __restrict__ rowmap) {
  int t = blockIdx.x * blockDim.x + threadIdx.x;
  if (t >= BATCH * NTOK) return;
  rowmap[t] = (t >> 8) * CLN + idxc[t];
}

__global__ void segw_kernel(const float* __restrict__ w, const int* __restrict__ idxc,
                            float* __restrict__ wsum) {
  int t = blockIdx.x * blockDim.x + threadIdx.x;
  if (t >= BATCH * NTOK) return;
  atomicAdd(&wsum[(t >> 8) * CLN + idxc[t]], w[t]);
}
__global__ void segx_kernel(const float* __restrict__ x, const float* __restrict__ w,
                            const int* __restrict__ idxc, float* __restrict__ xsum) {
  long total = (long)BATCH * NTOK * CD0;
  long t = (long)blockIdx.x * blockDim.x + threadIdx.x;
  long stride = (long)gridDim.x * blockDim.x;
  for (; t < total; t += stride) {
    long r = t / CD0; int c = (int)(t - r * CD0);
    int b = (int)(r >> 8);
    atomicAdd(&xsum[((long)(b * CLN + idxc[r])) * CD0 + c], x[t] * w[r]);
  }
}
__global__ void xm_kernel(const float* __restrict__ xsum, const float* __restrict__ wsum,
                          float* __restrict__ xm) {
  long total = (long)BATCH * CLN * CD0;
  long t = (long)blockIdx.x * blockDim.x + threadIdx.x;
  long stride = (long)gridDim.x * blockDim.x;
  for (; t < total; t += stride) {
    long cl = t / CD0;
    xm[t] = xsum[t] / (wsum[cl] + 1e-6f);
  }
}

// ---------------------------------------------------------------------------
// Attention-saliency map: am[b,m] = sum_{h<2} sum_n attn[b,h,n,m]; then min-max
// ---------------------------------------------------------------------------
__global__ void attnmap_kernel(const float* __restrict__ attn, float* __restrict__ am) {
  int t = blockIdx.x * blockDim.x + threadIdx.x;
  if (t >= BATCH * NTOK) return;
  int b = t >> 8, m = t & 255;
  float s = 0.f;
  for (int h = 0; h < 2; h++)
    for (int n = 0; n < NTOK; n++)
      s += attn[(((long)(b * H0_ + h) * NTOK) + n) * NTOK + m];
  am[t] = s;
}
__global__ __launch_bounds__(256)
void minmax_norm_kernel(const float* __restrict__ am, float* __restrict__ asout,
                        float* __restrict__ asmap_out) {
  __shared__ float smn[256], smx[256];
  int b = blockIdx.x, t = threadIdx.x;
  float v = am[b * NTOK + t];
  smn[t] = v; smx[t] = v; __syncthreads();
  for (int s = 128; s; s >>= 1) {
    if (t < s) { smn[t] = fminf(smn[t], smn[t + s]); smx[t] = fmaxf(smx[t], smx[t + s]); }
    __syncthreads();
  }
  float out = (v - smn[0]) / (smx[0] - smn[0]);
  asout[b * NTOK + t] = out;
  asmap_out[b * NTOK + t] = out;
}

// ===========================================================================
// Host-side orchestration
// ===========================================================================
struct BlockW {
  const float *ln1g, *ln1b, *qw, *kvw, *pw, *pb, *ln2g, *ln2b, *f1w, *f1b, *f2w, *f2b;
  const float *srw, *srb, *srg, *srbb;
};
static BlockW loadBlock(void* const* d, int i, bool sr) {
  BlockW w{};
  w.ln1g = (const float*)d[i + 0];  w.ln1b = (const float*)d[i + 1];
  w.qw   = (const float*)d[i + 2];  w.kvw  = (const float*)d[i + 3];
  w.pw   = (const float*)d[i + 4];  w.pb   = (const float*)d[i + 5];
  w.ln2g = (const float*)d[i + 6];  w.ln2b = (const float*)d[i + 7];
  w.f1w  = (const float*)d[i + 8];  w.f1b  = (const float*)d[i + 9];
  w.f2w  = (const float*)d[i + 10]; w.f2b  = (const float*)d[i + 11];
  if (sr) {
    w.srw = (const float*)d[i + 12]; w.srb  = (const float*)d[i + 13];
    w.srg = (const float*)d[i + 14]; w.srbb = (const float*)d[i + 15];
  }
  return w;
}

struct Ctx {
  float *X, *H, *Q, *KV, *ATTN, *MID;
  float *weight, *x2, *dens, *dpv, *am, *asout, *dmaxv, *wsum, *xsum, *xm, *kvtok, *xc;
  int   *index_down, *idxc, *rowmap;
};

static void gemm(hipStream_t s, const float* A, const float* Bm, float* C, const float* bias,
                 int M, int N, int K, long lda, long ldb, long ldc,
                 bool transB, int act,
                 int batch = 1, int innerB = 1,
                 long sAo = 0, long sAi = 0, long sBo = 0, long sBi = 0,
                 long sCo = 0, long sCi = 0,
                 const int* rowGather = nullptr)
{
  dim3 g((N + 31) / 32, (M + 127) / 128, batch);
  wmma_gemm_kernel<<<g, 256, 0, s>>>(A, Bm, C, bias, M, N, K, lda, ldb, ldc,
                                     transB ? 1 : 0, act, innerB,
                                     sAo, sAi, sBo, sBi, sCo, sCi, rowGather);
}

static void layernorm(hipStream_t s, const float* x, float* y,
                      const float* g, const float* b, long rows, int C) {
  ln_kernel<<<(unsigned)((rows + 7) / 8), 256, 0, s>>>(x, y, g, b, rows, C);
}
static void vadd(hipStream_t s, float* x, const float* a, long n) {
  vadd_kernel<<<4096, 256, 0, s>>>(x, a, n);
}

// One transformer block (pre-LN attention + MLP), residual in-place on x
static void block_forward(hipStream_t s, float* x, int Ntok, int C, int heads, int sr,
                          const BlockW& w, Ctx& c)
{
  const long rows = (long)BATCH * Ntok;
  const int  hd   = C / heads;
  layernorm(s, x, c.H, w.ln1g, w.ln1b, rows, C);

  const float* kvsrc;
  int M2;
  if (sr > 1) {                        // 8x8/stride8 conv token reduction -> 4 kv tokens
    im2col_sr_kernel<<<4096, 256, 0, s>>>(c.H, c.MID);
    gemm(s, c.MID, w.srw, c.kvtok, w.srb,
         BATCH * 4, C, C * 64, (long)C * 64, (long)C * 64, C, /*transB=*/true, 0);
    layernorm(s, c.kvtok, c.kvtok, w.srg, w.srbb, BATCH * 4, C);
    kvsrc = c.kvtok; M2 = 4;
  } else {
    kvsrc = c.H; M2 = Ntok;
  }

  // q = h @ q_w ; kv = kvsrc @ kv_w
  gemm(s, c.H, w.qw, c.Q, nullptr, (int)rows, C, C, C, C, C, false, 0);
  gemm(s, kvsrc, w.kvw, c.KV, nullptr, BATCH * M2, 2 * C, C, C, 2L * C, 2L * C, false, 0);

  // attn = q @ k^T  (per b,h)
  gemm(s, c.Q, c.KV, c.ATTN, nullptr,
       Ntok, M2, hd, C, 2L * C, M2, /*transB=*/true, 0,
       BATCH * heads, heads,
       (long)Ntok * C, hd, (long)M2 * 2 * C, hd,
       (long)heads * Ntok * M2, (long)Ntok * M2);
  softmax_kernel<<<(unsigned)(((long)BATCH * heads * Ntok + 7) / 8), 256, 0, s>>>(
      c.ATTN, (long)BATCH * heads * Ntok, M2, rsqrtf((float)hd));

  // out = attn @ v (per b,h) -> Q reused as [B,Ntok,C]
  gemm(s, c.ATTN, c.KV + C, c.Q, nullptr,
       Ntok, hd, M2, M2, 2L * C, C, false, 0,
       BATCH * heads, heads,
       (long)heads * Ntok * M2, (long)Ntok * M2,
       (long)M2 * 2 * C, hd,
       (long)Ntok * C, hd);

  // proj + residual
  gemm(s, c.Q, w.pw, c.H, w.pb, (int)rows, C, C, C, C, C, false, 0);
  vadd(s, x, c.H, rows * (long)C);

  // MLP
  layernorm(s, x, c.H, w.ln2g, w.ln2b, rows, C);
  gemm(s, c.H, w.f1w, c.MID, w.f1b, (int)rows, 4 * C, C, C, 4L * C, 4L * C, false, /*gelu*/1);
  gemm(s, c.MID, w.f2w, c.H, w.f2b, (int)rows, C, 4 * C, 4L * C, C, C, false, 0);
  vadd(s, x, c.H, rows * (long)C);
}

// CTM: density clustering + weighted token merge; writes xc [B,CLN,D1], idxc, rowmap
static void run_ctm(hipStream_t s, const float* x,
                    const float* sw, const float* sb, const float* pw, const float* pb,
                    Ctx& c)
{
  const long rows = (long)BATCH * NTOK;
  ctm_prep_kernel<<<(unsigned)((rows + 7) / 8), 256, 0, s>>>(x, sw, sb, c.asout, c.weight, c.x2,
                                                             rows, CD0);
  // gram = x @ x^T per batch (into ATTN region)
  gemm(s, x, x, c.ATTN, nullptr, NTOK, NTOK, CD0, CD0, CD0, NTOK, /*transB=*/true, 0,
       BATCH, 1, (long)NTOK * CD0, 0, (long)NTOK * CD0, 0, (long)NTOK * NTOK, 0);
  dist_kernel<<<4096, 256, 0, s>>>(c.ATTN, c.x2);
  density_kernel<<<(BATCH * NTOK + 255) / 256, 256, 0, s>>>(c.ATTN, c.dens);
  dmax_kernel<<<BATCH, 256, 0, s>>>(c.ATTN, c.dmaxv);
  dp_kernel<<<(BATCH * NTOK + 255) / 256, 256, 0, s>>>(c.ATTN, c.dens, c.dmaxv, c.dpv);
  top32_kernel<<<BATCH, 32, 0, s>>>(c.dpv, c.index_down);
  assign_kernel<<<(BATCH * NTOK + 255) / 256, 256, 0, s>>>(c.ATTN, c.index_down, c.idxc);
  override_kernel<<<(BATCH * CLN + 255) / 256, 256, 0, s>>>(c.index_down, c.idxc);
  rowmap_kernel<<<(BATCH * NTOK + 255) / 256, 256, 0, s>>>(c.idxc, c.rowmap);

  fz_kernel<<<64, 256, 0, s>>>(c.wsum, BATCH * CLN);
  fz_kernel<<<2048, 256, 0, s>>>(c.xsum, (long)BATCH * CLN * CD0);
  segw_kernel<<<(BATCH * NTOK + 255) / 256, 256, 0, s>>>(c.weight, c.idxc, c.wsum);
  segx_kernel<<<4096, 256, 0, s>>>(x, c.weight, c.idxc, c.xsum);
  xm_kernel<<<2048, 256, 0, s>>>(c.xsum, c.wsum, c.xm);

  // xc = xm @ proj + b : [B*CLN, D0] -> [B*CLN, D1]
  gemm(s, c.xm, pw, c.xc, pb, BATCH * CLN, CD1, CD0, CD0, CD1, CD1, false, 0);
}

// x += gather(xc, idxc) @ recover_w + recover_b   (rowGather fuses the gather)
static void run_recover(hipStream_t s, float* x, const float* rw, const float* rb, Ctx& c) {
  gemm(s, c.xc, rw, c.H, rb, BATCH * NTOK, CD0, CD1, CD1, CD0, CD0, false, 0,
       1, 1, 0, 0, 0, 0, 0, 0, c.rowmap);
  vadd(s, x, c.H, (long)BATCH * NTOK * CD0);
}

// ---------------------------------------------------------------------------
// Input index map (DFS over the reference setup_inputs() dict)
// ---------------------------------------------------------------------------
enum {
  IN_IMG = 0, IN_PATCH_W, IN_PATCH_B, IN_POS, IN_TP_W, IN_TP_B,
  BLK0AS_BASE = 6,            // 12 entries (no sr)
  IN_N0AS_G = 18, IN_N0AS_B,
  IN_CTMAS_SW = 20, IN_CTMAS_SB, IN_CTMAS_PW, IN_CTMAS_PB,
  BLK1AS_BASE = 24,           // 12 entries
  IN_N1AS_G = 36, IN_N1AS_B,
  IN_REC_W = 38, IN_REC_B,
  BLK0_BASE = 40,             // 2 x 16 entries (with sr)
  IN_N0_G = 72, IN_N0_B,
  IN_CTM1_SW = 74, IN_CTM1_SB, IN_CTM1_PW, IN_CTM1_PB,
  BLK1_BASE = 78,             // 2 x 12 entries
  IN_N1_G = 102, IN_N1_B,
  IN_CHG_W = 104, IN_CHG_B
};

extern "C" void kernel_launch(void* const* d_in, const int* in_sizes, int n_in,
                              void* d_out, int out_size, void* d_ws, size_t ws_size,
                              hipStream_t stream)
{
  (void)in_sizes; (void)n_in; (void)out_size; (void)ws_size;
  float* W = (float*)d_ws;

  // ---- workspace arena (float offsets) ----
  Ctx c{};
  size_t o = 0;
  c.X    = W + o; o += (size_t)BATCH * NTOK * CD0;          // 6.29M
  c.H    = W + o; o += (size_t)BATCH * NTOK * CD0;
  c.Q    = W + o; o += (size_t)BATCH * NTOK * CD1;          // big enough for C=768 blocks
  c.KV   = W + o; o += (size_t)BATCH * NTOK * 2 * CD1;      // kv [.,2C] worst case
  c.ATTN = W + o; o += (size_t)BATCH * H1_ * NTOK * NTOK;   // covers attn & gram/dist
  c.MID  = W + o; o += (size_t)BATCH * NTOK * 4 * CD1;      // fc1 out worst case / im2col
  c.weight = W + o; o += 16384;
  c.x2     = W + o; o += 16384;
  c.dens   = W + o; o += 16384;
  c.dpv    = W + o; o += 16384;
  c.am     = W + o; o += 16384;
  c.asout  = W + o; o += 16384;
  c.dmaxv  = W + o; o += 16384;
  c.wsum   = W + o; o += 16384;
  c.xsum   = W + o; o += (size_t)BATCH * CLN * CD0;
  c.xm     = W + o; o += (size_t)BATCH * CLN * CD0;
  c.kvtok  = W + o; o += (size_t)BATCH * 4 * CD0;
  c.xc     = W + o; o += (size_t)BATCH * CLN * CD1;
  c.index_down = (int*)(W + o); o += BATCH * CLN;
  c.idxc       = (int*)(W + o); o += BATCH * NTOK;
  c.rowmap     = (int*)(W + o); o += BATCH * NTOK;

  const float* img = (const float*)d_in[IN_IMG];
  float* out = (float*)d_out;
  float* asmap_out = out + (size_t)BATCH * NTOK * CD1;

  // ---- patch embedding (im2col + GEMM) + pos + to_patch ----
  im2col_patch_kernel<<<8192, 256, 0, stream>>>(img, c.MID);
  gemm(stream, c.MID, (const float*)d_in[IN_PATCH_W], c.KV, (const float*)d_in[IN_PATCH_B],
       BATCH * NTOK, CHID, 768, 768, 768, CHID, /*transB(OIHW)*/true, 0);
  addpos_kernel<<<8192, 256, 0, stream>>>(c.KV, (const float*)d_in[IN_POS],
                                          (long)BATCH * NTOK * CHID, (long)NTOK * CHID);
  gemm(stream, c.KV, (const float*)d_in[IN_TP_W], c.X, (const float*)d_in[IN_TP_B],
       BATCH * NTOK, CD0, CHID, CHID, CD0, CD0, false, 0);

  // ---- blk0_as (keeps attn for saliency map) ----
  BlockW b0as = loadBlock(d_in, BLK0AS_BASE, false);
  block_forward(stream, c.X, NTOK, CD0, H0_, 1, b0as, c);
  layernorm(stream, c.X, c.X, (const float*)d_in[IN_N0AS_G], (const float*)d_in[IN_N0AS_B],
            (long)BATCH * NTOK, CD0);

  attnmap_kernel<<<(BATCH * NTOK + 255) / 256, 256, 0, stream>>>(c.ATTN, c.am);
  minmax_norm_kernel<<<BATCH, 256, 0, stream>>>(c.am, c.asout, asmap_out);

  // ---- AS stage: ctm -> blk1_as -> norm -> recover ----
  run_ctm(stream, c.X, (const float*)d_in[IN_CTMAS_SW], (const float*)d_in[IN_CTMAS_SB],
          (const float*)d_in[IN_CTMAS_PW], (const float*)d_in[IN_CTMAS_PB], c);
  BlockW b1as = loadBlock(d_in, BLK1AS_BASE, false);
  block_forward(stream, c.xc, CLN, CD1, H1_, 1, b1as, c);
  layernorm(stream, c.xc, c.xc, (const float*)d_in[IN_N1AS_G], (const float*)d_in[IN_N1AS_B],
            (long)BATCH * CLN, CD1);
  run_recover(stream, c.X, (const float*)d_in[IN_REC_W], (const float*)d_in[IN_REC_B], c);

  // ---- main iterations ----
  BlockW blk0a = loadBlock(d_in, BLK0_BASE, true);
  BlockW blk0b = loadBlock(d_in, BLK0_BASE + 16, true);
  BlockW blk1a = loadBlock(d_in, BLK1_BASE, false);
  BlockW blk1b = loadBlock(d_in, BLK1_BASE + 12, false);
  for (int it = 0; it < NITER; it++) {
    block_forward(stream, c.X, NTOK, CD0, H0_, SR0_, blk0a, c);
    block_forward(stream, c.X, NTOK, CD0, H0_, SR0_, blk0b, c);
    layernorm(stream, c.X, c.X, (const float*)d_in[IN_N0_G], (const float*)d_in[IN_N0_B],
              (long)BATCH * NTOK, CD0);
    run_ctm(stream, c.X, (const float*)d_in[IN_CTM1_SW], (const float*)d_in[IN_CTM1_SB],
            (const float*)d_in[IN_CTM1_PW], (const float*)d_in[IN_CTM1_PB], c);
    block_forward(stream, c.xc, CLN, CD1, H1_, 1, blk1a, c);
    block_forward(stream, c.xc, CLN, CD1, H1_, 1, blk1b, c);
    layernorm(stream, c.xc, c.xc, (const float*)d_in[IN_N1_G], (const float*)d_in[IN_N1_B],
              (long)BATCH * CLN, CD1);
    run_recover(stream, c.X, (const float*)d_in[IN_REC_W], (const float*)d_in[IN_REC_B], c);
  }

  // ---- encoded = x @ change + b, straight into d_out ----
  gemm(stream, c.X, (const float*)d_in[IN_CHG_W], out, (const float*)d_in[IN_CHG_B],
       BATCH * NTOK, CD1, CD0, CD0, CD1, CD1, false, 0);
}
